// kNNHyperbolicAttentionLayer_83296595738917
// MI455X (gfx1250) — matrix operand order repaired and checked by reading, not compiled, not run on hardware
//
#include <hip/hip_runtime.h>
#include <hip/hip_bf16.h>
#include <math.h>

// ---------------- problem constants ----------------
constexpr int  CB    = 2;          // batch
constexpr int  CN    = 2048;       // sequence
constexpr int  CDIM  = 768;
constexpr int  CH    = 12;         // heads
constexpr int  CHD   = 64;         // head dim
constexpr int  CK    = 32;         // knn K (== wave32!)
constexpr int  CP    = 16;         // position dim
constexpr int  CM    = CB * CN;    // 4096 rows
constexpr int  CD4   = 4 * CDIM;   // 3072

constexpr int  KSUP  = 128;        // k-superstep staged by TDM
constexpr int  LPAD  = 8;          // f16 elements of LDS row padding (16B)
constexpr int  LROW  = KSUP + LPAD;            // 136 f16 per LDS row
constexpr int  LBUF  = 128 * LROW;             // elements per LDS buffer

typedef __attribute__((ext_vector_type(16))) _Float16 v16h;
typedef __attribute__((ext_vector_type(8)))  _Float16 v8h;
typedef __attribute__((ext_vector_type(8)))  float    v8f;
typedef __attribute__((ext_vector_type(4)))  unsigned int u32x4;
typedef __attribute__((ext_vector_type(8)))  int      i32x8;
typedef __attribute__((ext_vector_type(4)))  int      i32x4;

#if defined(__has_builtin)
#if __has_builtin(__builtin_amdgcn_tensor_load_to_lds)
#define HAVE_TDM 1
#endif
#endif
#ifndef HAVE_TDM
#define HAVE_TDM 0
#endif

// ---------------- f32 -> f16 convert ----------------
__global__ void cvt_f32_f16(const float* __restrict__ src, _Float16* __restrict__ dst, int n) {
    int i = blockIdx.x * blockDim.x + threadIdx.x;
    if (i < n) dst[i] = (_Float16)src[i];
}

// ---------------- LayerNorm (wave per row) -> f16 ----------------
__global__ void layernorm_f16(const float* __restrict__ x, const float* __restrict__ g,
                              const float* __restrict__ be, _Float16* __restrict__ out,
                              int rows, int dim) {
    int wid  = (blockIdx.x * blockDim.x + threadIdx.x) >> 5;
    int lane = threadIdx.x & 31;
    if (wid >= rows) return;
    const float* xr = x + (size_t)wid * dim;
    float s = 0.f, s2 = 0.f;
    for (int d = lane; d < dim; d += 32) { float v = xr[d]; s += v; s2 += v * v; }
    for (int m = 16; m > 0; m >>= 1) { s += __shfl_xor(s, m, 32); s2 += __shfl_xor(s2, m, 32); }
    float mean = s / dim;
    float var  = s2 / dim - mean * mean;
    float inv  = rsqrtf(var + 1e-5f);
    _Float16* orow = out + (size_t)wid * dim;
    for (int d = lane; d < dim; d += 32)
        orow[d] = (_Float16)((xr[d] - mean) * inv * g[d] + be[d]);
}

// ---------------- position squared norms ----------------
__global__ void sumsq_pos(const float* __restrict__ pos, float* __restrict__ xs, int rows) {
    int i = blockIdx.x * blockDim.x + threadIdx.x;
    if (i >= rows) return;
    const float* p = pos + (size_t)i * CP;
    float s = 0.f;
#pragma unroll
    for (int j = 0; j < CP; ++j) s += p[j] * p[j];
    xs[i] = s;
}

// ---------------- TDM issue: stage W[n0b..n0b+128) x [k0s..k0s+128) into LDS ----------------
__device__ inline void tdm_stage_tile(const _Float16* W, int n0b, int k0s, int Kd, int Ncols,
                                      unsigned ldsOff) {
#if HAVE_TDM
    unsigned long long ga = (unsigned long long)(uintptr_t)W +
                            ((unsigned long long)n0b * (unsigned)Kd + (unsigned)k0s) * 2ull;
    u32x4 g0;
    g0.x = 1u;                                         // count=1, user mode, no gather
    g0.y = ldsOff;                                     // LDS byte address of tile start
    g0.z = (unsigned)(ga & 0xFFFFFFFFull);             // global_addr[31:0]
    g0.w = (unsigned)((ga >> 32) & 0x1FFFFFFull) | (2u << 30);  // addr[56:32] | type=2

    i32x8 g1;
    // data_size=2B (code 1) @17:16, pad_enable @20, pad_interval=64DW (code 5) @24:22,
    // pad_amount=4DW (code 3) @31:25
    g1[0] = (1 << 16) | (1 << 20) | (5 << 22) | (3 << 25);
    g1[1] = (Kd & 0xFFFF) << 16;                       // tensor_dim0 lo16 @ bits 79:48
    g1[2] = ((Kd >> 16) & 0xFFFF) | ((Ncols & 0xFFFF) << 16);   // dim0 hi | dim1 lo
    g1[3] = ((Ncols >> 16) & 0xFFFF) | (KSUP << 16);   // dim1 hi | tile_dim0=128
    g1[4] = 128;                                       // tile_dim1=128, tile_dim2=0
    g1[5] = Kd;                                        // tensor_dim0_stride lo32
    g1[6] = (Kd & 0xFFFF) << 16;                       // stride0 hi16=0 | stride1 lo16
    g1[7] = (Kd >> 16) & 0xFFFF;                       // stride1 hi

    i32x4 z4 = (i32x4){0, 0, 0, 0};
#if __clang_major__ >= 23
    i32x8 z8 = (i32x8){0, 0, 0, 0, 0, 0, 0, 0};
    __builtin_amdgcn_tensor_load_to_lds(g0, g1, z4, z4, z8, 0);
#else
    __builtin_amdgcn_tensor_load_to_lds(g0, g1, z4, z4, 0);
#endif
#endif
}

// ---------------- WMMA GEMM with TDM-staged weight tiles ----------------
// C[M,Ncols] = A[M,Kd](f16) * W[Ncols,Kd]^T + bias
// EPI: 1 = store f16; 2 = exact-GELU then f16; 3 = +residual(f32), store f32
template<int EPI>
__global__ __launch_bounds__(256)
void gemm_wmma(const _Float16* __restrict__ A, const _Float16* __restrict__ W,
               const float* __restrict__ bias, const float* __restrict__ res,
               float* __restrict__ outF, _Float16* __restrict__ outH,
               int M, int Ncols, int Kd) {
    __shared__ _Float16 smem[2][LBUF];                 // 2 x 128 cols x (128+8) f16

    const int lane  = threadIdx.x & 31;
    const int wave  = threadIdx.x >> 5;
    const int wm    = wave >> 2;                       // 0..1 (M dir)
    const int wn    = wave & 3;                        // 0..3 (N dir)
    const int m0    = blockIdx.y * 128 + wm * 64;
    const int n0b   = blockIdx.x * 128;                // block col base
    const int laneM = lane & 15;
    const int kselA = (lane >> 4) << 3;                // lanes 0-15: +0 ; 16-31: +8
    const int kselB = (lane >> 4) << 3;                // lanes 0-15: +0 ; 16-31: +8

    v8f acc[4][2];
#pragma unroll
    for (int i = 0; i < 4; ++i)
#pragma unroll
        for (int j = 0; j < 2; ++j) acc[i][j] = (v8f){0.f,0.f,0.f,0.f,0.f,0.f,0.f,0.f};

    const int S = Kd / KSUP;                           // k-supersteps

#if HAVE_TDM
    if (wave == 0)                                     // prologue: stage superstep 0
        tdm_stage_tile(W, n0b, 0, Kd, Ncols, 0u);
#endif

    for (int s = 0; s < S; ++s) {
        const int buf = s & 1;
        const int k0s = s * KSUP;

#if HAVE_TDM
        if (wave == 0) __builtin_amdgcn_s_wait_tensorcnt(0);
        __syncthreads();                               // tile[buf] visible to all waves
        if (wave == 0 && (s + 1) < S)                  // prefetch next tile into other buffer
            tdm_stage_tile(W, n0b, k0s + KSUP, Kd, Ncols, (unsigned)((buf ^ 1) * LBUF * 2));
#else
        // fallback: cooperative global->LDS copy of W tile (col-major rows of KSUP)
        {
            int t = threadIdx.x;                       // 256 threads, each copies 64 f16
            int col   = t >> 1;
            int khalf = (t & 1) * 64;
            const _Float16* src = W + (size_t)(n0b + col) * Kd + k0s + khalf;
            _Float16* dst = &smem[buf][col * LROW + khalf];
#pragma unroll
            for (int e = 0; e < 8; ++e)
                *(v8h*)(dst + 8 * e) = *(const v8h*)(src + 8 * e);
        }
        __syncthreads();
#endif

        for (int kk = 0; kk < KSUP; kk += 32) {
            v16h a[4], b[2];
#pragma unroll
            for (int i = 0; i < 4; ++i) {
                const _Float16* p = A + (size_t)(m0 + 16 * i + laneM) * Kd + k0s + kk + kselA;
                v8h lo = *(const v8h*)(p);
                v8h hi = *(const v8h*)(p + 16);
#pragma unroll
                for (int e = 0; e < 8; ++e) { a[i][e] = lo[e]; a[i][8 + e] = hi[e]; }
            }
#pragma unroll
            for (int j = 0; j < 2; ++j) {
                int colLocal = wn * 32 + 16 * j + laneM;
                const _Float16* p = &smem[buf][colLocal * LROW + kk + kselB];
                v8h lo = *(const v8h*)(p);
                v8h hi = *(const v8h*)(p + 16);
#pragma unroll
                for (int e = 0; e < 8; ++e) { b[j][e] = lo[e]; b[j][8 + e] = hi[e]; }
            }
#pragma unroll
            for (int i = 0; i < 4; ++i)
#pragma unroll
                for (int j = 0; j < 2; ++j)
                    acc[i][j] = __builtin_amdgcn_wmma_f32_16x16x32_f16(
                        false, a[i], false, b[j], (short)0, acc[i][j], false, false);
        }
        __syncthreads();                               // done reading tile[buf]
    }

    const int rsel = (lane >> 4) << 3;                 // C rows: lanes 16-31 hold M+8
#pragma unroll
    for (int i = 0; i < 4; ++i)
#pragma unroll
        for (int j = 0; j < 2; ++j)
#pragma unroll
            for (int v = 0; v < 8; ++v) {
                int rr = m0 + 16 * i + v + rsel;
                int cc = n0b + wn * 32 + 16 * j + laneM;
                float val = acc[i][j][v] + bias[cc];
                size_t o = (size_t)rr * Ncols + cc;
                if (EPI == 1) {
                    outH[o] = (_Float16)val;
                } else if (EPI == 2) {
                    outH[o] = (_Float16)(0.5f * val * (1.0f + erff(val * 0.70710678118f)));
                } else {
                    outF[o] = val + res[o];
                }
            }
}

// ---------------- Poincare distances + top-K selection (block per row) ----------------
__global__ __launch_bounds__(256)
void knn_topk(const float* __restrict__ pos, const float* __restrict__ xs,
              const float* __restrict__ cptr, float* __restrict__ dists,
              int* __restrict__ idxo) {
    __shared__ float sdist[CN];
    __shared__ float sval[256];
    __shared__ int   sidx[256];
    __shared__ float sp[CP];

    const int row = blockIdx.x;      // b*N + n
    const int b   = row / CN;
    const int tid = threadIdx.x;
    const float c0 = cptr[0];
    const float invsqc = rsqrtf(c0);

    if (tid < CP) sp[tid] = pos[(size_t)row * CP + tid];
    __syncthreads();

    const float xsn = xs[row];
    for (int m = tid; m < CN; m += 256) {
        const float* pm = pos + ((size_t)(b * CN + m)) * CP;
        float dot = 0.f;
#pragma unroll
        for (int j = 0; j < CP; ++j) dot += sp[j] * pm[j];
        float xsm = xs[b * CN + m];
        float sq  = fmaxf(xsn + xsm - 2.f * dot, 0.f);
        float den = (1.f - c0 * xsn) * (1.f - c0 * xsm) + 1e-8f;
        float arg = fmaxf(1.f + 2.f * c0 * sq / den, 1.f);
        sdist[m]  = acoshf(arg) * invsqc;
    }
    __syncthreads();

    for (int t = 0; t < CK; ++t) {
        float best = 3.0e38f; int bi = CN;
        for (int m = tid; m < CN; m += 256) {
            float v = sdist[m];
            if (v < best) { best = v; bi = m; }
        }
        sval[tid] = best; sidx[tid] = bi;
        __syncthreads();
        for (int s = 128; s > 0; s >>= 1) {
            if (tid < s) {
                float v2 = sval[tid + s]; int i2 = sidx[tid + s];
                if (v2 < sval[tid] || (v2 == sval[tid] && i2 < sidx[tid])) {
                    sval[tid] = v2; sidx[tid] = i2;
                }
            }
            __syncthreads();
        }
        if (tid == 0) {
            dists[(size_t)row * CK + t] = sval[0];
            idxo [(size_t)row * CK + t] = sidx[0];
            sdist[sidx[0]] = 3.0e38f;
        }
        __syncthreads();
    }
}

// ---------------- kNN attention: wave per (b,n,h), lane = neighbor (K==32) ----------------
__global__ __launch_bounds__(256)
void knn_attn(const _Float16* __restrict__ qh, const _Float16* __restrict__ kh,
              const _Float16* __restrict__ vh, const float* __restrict__ dists,
              const int* __restrict__ idx, const float* __restrict__ log_tau,
              _Float16* __restrict__ ao) {
    int wid  = (blockIdx.x * blockDim.x + threadIdx.x) >> 5;
    int lane = threadIdx.x & 31;
    if (wid >= CM * CH) return;
    int h   = wid % CH;
    int row = wid / CH;          // b*N + n
    int bN  = (row / CN) * CN;   // b*N

    const float tau = __expf(log_tau[0]) + 1e-8f;

    int   myIdx  = idx  [(size_t)row * CK + lane];
    float myDist = dists[(size_t)row * CK + lane];

    const _Float16* qp = qh + (size_t)row * CDIM + h * CHD;
    const _Float16* kp = kh + ((size_t)(bN + myIdx)) * CDIM + h * CHD;

    float dot = 0.f;
#pragma unroll 8
    for (int d = 0; d < CHD; ++d) dot += (float)qp[d] * (float)kp[d];
    float logit = dot * 0.125f - myDist / tau;   // 1/sqrt(64) = 0.125

    float mx = logit;
    for (int m = 16; m > 0; m >>= 1) mx = fmaxf(mx, __shfl_xor(mx, m, 32));
    float e = __expf(logit - mx);
    float s = e;
    for (int m = 16; m > 0; m >>= 1) s += __shfl_xor(s, m, 32);
    float p = e / s;

    float acc0 = 0.f, acc1 = 0.f;
    for (int k = 0; k < CK; ++k) {
        float pk = __shfl(p, k, 32);
        int   ik = __shfl(myIdx, k, 32);
        const _Float16* vp = vh + ((size_t)(bN + ik)) * CDIM + h * CHD;
        acc0 += pk * (float)vp[lane];
        acc1 += pk * (float)vp[lane + 32];
    }
    _Float16* op = ao + (size_t)row * CDIM + h * CHD;
    op[lane]      = (_Float16)acc0;
    op[lane + 32] = (_Float16)acc1;
}

// ---------------- host launch ----------------
extern "C" void kernel_launch(void* const* d_in, const int* in_sizes, int n_in,
                              void* d_out, int out_size, void* d_ws, size_t ws_size,
                              hipStream_t stream) {
    const float* x       = (const float*)d_in[0];
    const float* pos     = (const float*)d_in[1];
    const float* c       = (const float*)d_in[2];
    const float* Wq      = (const float*)d_in[3];
    const float* bq      = (const float*)d_in[4];
    const float* Wk      = (const float*)d_in[5];
    const float* bk      = (const float*)d_in[6];
    const float* Wv      = (const float*)d_in[7];
    const float* bv      = (const float*)d_in[8];
    const float* Wo      = (const float*)d_in[9];
    const float* bo      = (const float*)d_in[10];
    const float* W1      = (const float*)d_in[11];
    const float* b1      = (const float*)d_in[12];
    const float* W2      = (const float*)d_in[13];
    const float* b2      = (const float*)d_in[14];
    const float* g1      = (const float*)d_in[15];
    const float* be1     = (const float*)d_in[16];
    const float* g2      = (const float*)d_in[17];
    const float* be2     = (const float*)d_in[18];
    const float* log_tau = (const float*)d_in[19];
    float* out = (float*)d_out;

    // workspace carve-up (256B aligned)
    char* w = (char*)d_ws;
    size_t off = 0;
    auto alloc = [&](size_t bytes) { void* p = w + off; off = (off + bytes + 255) & ~(size_t)255; return p; };
    _Float16* xn_h = (_Float16*)alloc((size_t)CM * CDIM * 2);
    _Float16* Wq_h = (_Float16*)alloc((size_t)CDIM * CDIM * 2);
    _Float16* Wk_h = (_Float16*)alloc((size_t)CDIM * CDIM * 2);
    _Float16* Wv_h = (_Float16*)alloc((size_t)CDIM * CDIM * 2);
    _Float16* Wo_h = (_Float16*)alloc((size_t)CDIM * CDIM * 2);
    _Float16* W1_h = (_Float16*)alloc((size_t)CD4 * CDIM * 2);
    _Float16* W2_h = (_Float16*)alloc((size_t)CDIM * CD4 * 2);
    _Float16* q_h  = (_Float16*)alloc((size_t)CM * CDIM * 2);
    _Float16* k_h  = (_Float16*)alloc((size_t)CM * CDIM * 2);
    _Float16* v_h  = (_Float16*)alloc((size_t)CM * CDIM * 2);
    _Float16* ao_h = (_Float16*)alloc((size_t)CM * CDIM * 2);
    float*    x2_f = (float*)   alloc((size_t)CM * CDIM * 4);
    _Float16* h_h  = (_Float16*)alloc((size_t)CM * CDIM * 2);
    _Float16* h1_h = (_Float16*)alloc((size_t)CM * CD4 * 2);
    float*    xs_f = (float*)   alloc((size_t)CM * 4);
    float*    d_f  = (float*)   alloc((size_t)CM * CK * 4);
    int*      i_i  = (int*)     alloc((size_t)CM * CK * 4);
    (void)ws_size; (void)n_in; (void)in_sizes; (void)out_size;

    // 1) weights -> f16
    auto cvt = [&](const float* s, _Float16* d, int n) {
        cvt_f32_f16<<<(n + 255) / 256, 256, 0, stream>>>(s, d, n);
    };
    cvt(Wq, Wq_h, CDIM * CDIM);
    cvt(Wk, Wk_h, CDIM * CDIM);
    cvt(Wv, Wv_h, CDIM * CDIM);
    cvt(Wo, Wo_h, CDIM * CDIM);
    cvt(W1, W1_h, CD4 * CDIM);
    cvt(W2, W2_h, CDIM * CD4);

    // 2) LN1 (8 waves/block, wave per row)
    layernorm_f16<<<CM / 8, 256, 0, stream>>>(x, g1, be1, xn_h, CM, CDIM);

    // 3) q,k,v projections (WMMA + TDM-staged weights, f16 out)
    dim3 gP(CDIM / 128, CM / 128);
    gemm_wmma<1><<<gP, 256, 0, stream>>>(xn_h, Wq_h, bq, nullptr, nullptr, q_h, CM, CDIM, CDIM);
    gemm_wmma<1><<<gP, 256, 0, stream>>>(xn_h, Wk_h, bk, nullptr, nullptr, k_h, CM, CDIM, CDIM);
    gemm_wmma<1><<<gP, 256, 0, stream>>>(xn_h, Wv_h, bv, nullptr, nullptr, v_h, CM, CDIM, CDIM);

    // 4) kNN geometry
    sumsq_pos<<<(CM + 255) / 256, 256, 0, stream>>>(pos, xs_f, CM);
    knn_topk<<<CM, 256, 0, stream>>>(pos, xs_f, c, d_f, i_i);

    // 5) attention (wave per (row, head))
    int nwaves = CM * CH;
    knn_attn<<<(nwaves * 32 + 255) / 256, 256, 0, stream>>>(q_h, k_h, v_h, d_f, i_i, log_tau, ao_h);

    // 6) x2 = x + ao @ Wo^T + bo  (f32 out)
    gemm_wmma<3><<<gP, 256, 0, stream>>>(ao_h, Wo_h, bo, x, x2_f, nullptr, CM, CDIM, CDIM);

    // 7) LN2
    layernorm_f16<<<CM / 8, 256, 0, stream>>>(x2_f, g2, be2, h_h, CM, CDIM);

    // 8) FFN1 with exact GELU (f16 out)
    dim3 gF1(CD4 / 128, CM / 128);
    gemm_wmma<2><<<gF1, 256, 0, stream>>>(h_h, W1_h, b1, nullptr, nullptr, h1_h, CM, CD4, CDIM);

    // 9) out = x2 + gelu @ W2^T + b2
    gemm_wmma<3><<<gP, 256, 0, stream>>>(h1_h, W2_h, b2, x2_f, out, nullptr, CM, CDIM, CD4);
}
